// SNNAdapter_68899865362577
// MI455X (gfx1250) — compile-verified, matrix operand order checked by reading, not run on hardware
//
#include <hip/hip_runtime.h>

// ---------------------------------------------------------------------------
// MI455X (gfx1250, wave32):
//   x_proj = x @ w_in + b_in
//   spikes = LIF-scan_T(x_proj)          (4 graded thresholds, soft reset)
//   out    = x + alpha * (spikes @ w_out + b_out)
// GEMMs: BF16 WMMA (v_wmma_f32_16x16x32_bf16), f32 accumulate, double-buffered
// LDS staging with packed ds_store_b64 transpose writes.
// ---------------------------------------------------------------------------

typedef __attribute__((ext_vector_type(4)))  __bf16 v4bf;
typedef __attribute__((ext_vector_type(8)))  __bf16 v8bf;
typedef __attribute__((ext_vector_type(16))) __bf16 v16bf;
typedef __attribute__((ext_vector_type(8)))  float  v8f;

#define TB_M 128
#define TB_N 128
#define TB_K 32
#define LDSS 40   // bf16 elems per row: 80 B stride -> near-optimal ds_load_b128

// Fragment: slots 0..7 = K kbase..kbase+7, slots 8..15 = K kbase+16..kbase+23.
__device__ __forceinline__ v16bf make_frag(const __bf16* p0, const __bf16* p1) {
    v8bf lo = *(const v8bf*)p0;
    v8bf hi = *(const v8bf*)p1;
    return __builtin_shufflevector(lo, hi, 0,1,2,3,4,5,6,7,8,9,10,11,12,13,14,15);
}

// 8 WMMAs on a 64x32 wave patch from the current LDS buffers.
__device__ __forceinline__ void wmma_tile(const __bf16 (*A)[LDSS],
                                          const __bf16 (*B)[LDSS],
                                          int warpM, int warpN, int lm, int kbase,
                                          v8f acc[4][2]) {
    v16bf afr[4], bfr[2];
    #pragma unroll
    for (int mi = 0; mi < 4; ++mi) {
        const int m = warpM * 64 + mi * 16 + lm;
        afr[mi] = make_frag(&A[m][kbase], &A[m][kbase + 16]);
    }
    #pragma unroll
    for (int ni = 0; ni < 2; ++ni) {
        const int n = warpN * 32 + ni * 16 + lm;
        bfr[ni] = make_frag(&B[n][kbase], &B[n][kbase + 16]);
    }
    #pragma unroll
    for (int mi = 0; mi < 4; ++mi)
        #pragma unroll
        for (int ni = 0; ni < 2; ++ni)
            acc[mi][ni] = __builtin_amdgcn_wmma_f32_16x16x32_bf16(
                false, afr[mi], false, bfr[ni], (short)0, acc[mi][ni], false, false);
}

// B-tile (f32 [K,N] -> bf16 LDS [n][k]) 4x4 micro-tile transpose staging.
__device__ __forceinline__ void b_load(const float* __restrict__ W, int N,
                                       int kb, int N0, int k0, int n0, float4 bv[4]) {
    #pragma unroll
    for (int i = 0; i < 4; ++i)
        bv[i] = *(const float4*)(W + (size_t)(kb + k0 + i) * N + N0 + n0);
}
__device__ __forceinline__ void b_store(__bf16 (*B)[LDSS], int k0, int n0,
                                        const float4 bv[4]) {
    v4bf t0 = {(__bf16)bv[0].x, (__bf16)bv[1].x, (__bf16)bv[2].x, (__bf16)bv[3].x};
    v4bf t1 = {(__bf16)bv[0].y, (__bf16)bv[1].y, (__bf16)bv[2].y, (__bf16)bv[3].y};
    v4bf t2 = {(__bf16)bv[0].z, (__bf16)bv[1].z, (__bf16)bv[2].z, (__bf16)bv[3].z};
    v4bf t3 = {(__bf16)bv[0].w, (__bf16)bv[1].w, (__bf16)bv[2].w, (__bf16)bv[3].w};
    *(v4bf*)&B[n0 + 0][k0] = t0;
    *(v4bf*)&B[n0 + 1][k0] = t1;
    *(v4bf*)&B[n0 + 2][k0] = t2;
    *(v4bf*)&B[n0 + 3][k0] = t3;
}

// ------------------------- GEMM1: Y = X(f32) @ W(f32) + bias -----------------
__global__ __launch_bounds__(256) void snn_gemm1_kernel(
    const float* __restrict__ X,   // [M, K]
    const float* __restrict__ W,   // [K, N]
    const float* __restrict__ bias,// [N]
    float* __restrict__ Y,         // [M, N]
    int M, int N, int K)
{
    __shared__ __attribute__((aligned(16))) __bf16 Alds[2][TB_M][LDSS];
    __shared__ __attribute__((aligned(16))) __bf16 Blds[2][TB_N][LDSS];

    const int tid   = threadIdx.x;
    const int lane  = tid & 31;
    const int wave  = tid >> 5;
    const int warpM = wave >> 2;     // 0..1
    const int warpN = wave & 3;      // 0..3
    const int half  = lane >> 4;
    const int lm    = lane & 15;
    const int kbase = half * 8;

    const int M0 = blockIdx.y * TB_M;
    const int N0 = blockIdx.x * TB_N;

    // A loader: 32 rows/pass x 4 passes, float4 per thread
    const int a_tr = tid >> 3;           // 0..31
    const int a_tc = (tid & 7) << 2;     // 0..28
    // B loader: 4x4 micro-tile transpose, whole tile in one pass
    const int k0 = (tid >> 5) << 2;      // 0,4,..,28
    const int n0 = (tid & 31) << 2;      // 0,4,..,124

    v8f acc[4][2];
    #pragma unroll
    for (int mi = 0; mi < 4; ++mi)
        #pragma unroll
        for (int ni = 0; ni < 2; ++ni)
            acc[mi][ni] = (v8f)(0.0f);

    float4 av[4], bv[4];

    // ---- prologue: stage tile kb=0 into buffer 0
    #pragma unroll
    for (int p = 0; p < 4; ++p)
        av[p] = *(const float4*)(X + (size_t)(M0 + p * 32 + a_tr) * K + 0 + a_tc);
    b_load(W, N, 0, N0, k0, n0, bv);
    #pragma unroll
    for (int p = 0; p < 4; ++p) {
        v4bf t = {(__bf16)av[p].x, (__bf16)av[p].y, (__bf16)av[p].z, (__bf16)av[p].w};
        *(v4bf*)&Alds[0][p * 32 + a_tr][a_tc] = t;
    }
    b_store(Blds[0], k0, n0, bv);
    __syncthreads();

    int buf = 0;
    for (int kb = 0; kb < K; kb += TB_K) {
        const bool has_next = (kb + TB_K) < K;
        if (has_next) {   // issue next tile's global loads before the math
            const int kn = kb + TB_K;
            #pragma unroll
            for (int p = 0; p < 4; ++p)
                av[p] = *(const float4*)(X + (size_t)(M0 + p * 32 + a_tr) * K + kn + a_tc);
            b_load(W, N, kn, N0, k0, n0, bv);
        }

        wmma_tile(Alds[buf], Blds[buf], warpM, warpN, lm, kbase, acc);

        if (has_next) {   // convert + store while WMMAs drain
            #pragma unroll
            for (int p = 0; p < 4; ++p) {
                v4bf t = {(__bf16)av[p].x, (__bf16)av[p].y, (__bf16)av[p].z, (__bf16)av[p].w};
                *(v4bf*)&Alds[buf ^ 1][p * 32 + a_tr][a_tc] = t;
            }
            b_store(Blds[buf ^ 1], k0, n0, bv);
        }
        __syncthreads();
        buf ^= 1;
    }

    // Epilogue: Y = acc + bias   (C/D layout: N = lm, M = r + half*8)
    #pragma unroll
    for (int mi = 0; mi < 4; ++mi) {
        #pragma unroll
        for (int ni = 0; ni < 2; ++ni) {
            const int col = N0 + warpN * 32 + ni * 16 + lm;
            const float b = bias[col];
            #pragma unroll
            for (int r = 0; r < 8; ++r) {
                const int row = M0 + warpM * 64 + mi * 16 + half * 8 + r;
                Y[(size_t)row * N + col] = acc[mi][ni][r] + b;
            }
        }
    }
}

// ---------------- LIF scan over T (sequential per (b,h) chain) ---------------
__global__ __launch_bounds__(256) void snn_scan_kernel(
    const float* __restrict__ Xp,   // [B, T, H]
    __bf16* __restrict__ S,         // [B, T, H] graded spikes in {0..4}
    const float* __restrict__ thre_p,
    int T, int H)
{
    const int idx = blockIdx.x * blockDim.x + threadIdx.x; // b*H + h
    const int b = idx / H;
    const int h = idx - b * H;
    const float th = *thre_p;
    const float th2 = 2.0f * th, th3 = 3.0f * th, th4 = 4.0f * th;

    const float* __restrict__ p  = Xp + (size_t)b * T * H + h;
    __bf16* __restrict__      sp = S  + (size_t)b * T * H + h;

    float mem = 0.0f;
    for (int t = 0; t < T; t += 8) {
        float xs[8];
        #pragma unroll
        for (int j = 0; j < 8; ++j)          // batch strided loads: 8 in flight
            xs[j] = p[(size_t)(t + j) * H];
        #pragma unroll
        for (int j = 0; j < 8; ++j) {
            mem = 0.25f * mem + xs[j];
            const float s = (float)(mem >= th) + (float)(mem >= th2)
                          + (float)(mem >= th3) + (float)(mem >= th4);
            mem -= s * th;                       // soft reset
            sp[(size_t)(t + j) * H] = (__bf16)s; // exact in bf16 (0..4)
        }
    }
}

// ------- GEMM2: out = Xin + alpha * (S(bf16) @ W(f32->bf16) + bias) ----------
__global__ __launch_bounds__(256) void snn_gemm2_kernel(
    const __bf16* __restrict__ S,   // [M, K] spikes
    const float* __restrict__ W,    // [K, N]
    const float* __restrict__ bias, // [N]
    const float* __restrict__ Xin,  // [M, N] residual
    const float* __restrict__ alpha_p,
    float* __restrict__ Out,        // [M, N]
    int M, int N, int K)
{
    __shared__ __attribute__((aligned(16))) __bf16 Alds[2][TB_M][LDSS];
    __shared__ __attribute__((aligned(16))) __bf16 Blds[2][TB_N][LDSS];

    const int tid   = threadIdx.x;
    const int lane  = tid & 31;
    const int wave  = tid >> 5;
    const int warpM = wave >> 2;
    const int warpN = wave & 3;
    const int half  = lane >> 4;
    const int lm    = lane & 15;
    const int kbase = half * 8;

    const int M0 = blockIdx.y * TB_M;
    const int N0 = blockIdx.x * TB_N;

    // A loader (bf16 source): 64 rows/pass x 2 passes, 16B per thread
    const int a_tr = tid >> 2;           // 0..63
    const int a_tc = (tid & 3) << 3;     // 0,8,16,24
    const int k0 = (tid >> 5) << 2;
    const int n0 = (tid & 31) << 2;

    v8f acc[4][2];
    #pragma unroll
    for (int mi = 0; mi < 4; ++mi)
        #pragma unroll
        for (int ni = 0; ni < 2; ++ni)
            acc[mi][ni] = (v8f)(0.0f);

    v8bf sa[2];
    float4 bv[4];

    // ---- prologue: stage tile kb=0 into buffer 0
    #pragma unroll
    for (int p = 0; p < 2; ++p)
        sa[p] = *(const v8bf*)(S + (size_t)(M0 + p * 64 + a_tr) * K + 0 + a_tc);
    b_load(W, N, 0, N0, k0, n0, bv);
    #pragma unroll
    for (int p = 0; p < 2; ++p)
        *(v8bf*)&Alds[0][p * 64 + a_tr][a_tc] = sa[p];
    b_store(Blds[0], k0, n0, bv);
    __syncthreads();

    int buf = 0;
    for (int kb = 0; kb < K; kb += TB_K) {
        const bool has_next = (kb + TB_K) < K;
        if (has_next) {
            const int kn = kb + TB_K;
            #pragma unroll
            for (int p = 0; p < 2; ++p)
                sa[p] = *(const v8bf*)(S + (size_t)(M0 + p * 64 + a_tr) * K + kn + a_tc);
            b_load(W, N, kn, N0, k0, n0, bv);
        }

        wmma_tile(Alds[buf], Blds[buf], warpM, warpN, lm, kbase, acc);

        if (has_next) {
            #pragma unroll
            for (int p = 0; p < 2; ++p)
                *(v8bf*)&Alds[buf ^ 1][p * 64 + a_tr][a_tc] = sa[p];
            b_store(Blds[buf ^ 1], k0, n0, bv);
        }
        __syncthreads();
        buf ^= 1;
    }

    // Fused epilogue: out = x + alpha * (acc + b_out)
    const float alpha = *alpha_p;
    #pragma unroll
    for (int mi = 0; mi < 4; ++mi) {
        #pragma unroll
        for (int ni = 0; ni < 2; ++ni) {
            const int col = N0 + warpN * 32 + ni * 16 + lm;
            const float b = bias[col];
            #pragma unroll
            for (int r = 0; r < 8; ++r) {
                const int row = M0 + warpM * 64 + mi * 16 + half * 8 + r;
                const size_t o = (size_t)row * N + col;
                Out[o] = Xin[o] + alpha * (acc[mi][ni][r] + b);
            }
        }
    }
}

// ---------------------------------------------------------------------------
extern "C" void kernel_launch(void* const* d_in, const int* in_sizes, int n_in,
                              void* d_out, int out_size, void* d_ws, size_t ws_size,
                              hipStream_t stream) {
    (void)n_in; (void)out_size; (void)ws_size;
    const float* x     = (const float*)d_in[0]; // [B, T, H]
    const float* alpha = (const float*)d_in[1]; // scalar
    const float* thre  = (const float*)d_in[2]; // scalar
    const float* w_in  = (const float*)d_in[3]; // [H, H]
    const float* b_in  = (const float*)d_in[4]; // [H]
    const float* w_out = (const float*)d_in[5]; // [H, H]
    const float* b_out = (const float*)d_in[6]; // [H]
    float* out = (float*)d_out;

    const int H = in_sizes[4];          // 2048
    const int T = 2048;
    const int B = in_sizes[0] / (T * H);
    const int M = B * T;                // 8192

    float*  xproj  = (float*)d_ws;                                   // [M, H] f32
    __bf16* spikes = (__bf16*)((char*)d_ws + (size_t)M * H * sizeof(float));

    dim3 block(256);
    dim3 grid(H / TB_N, M / TB_M);

    snn_gemm1_kernel<<<grid, block, 0, stream>>>(x, w_in, b_in, xproj, M, H, H);
    snn_scan_kernel<<<(B * H) / 256, block, 0, stream>>>(xproj, spikes, thre, T, H);
    snn_gemm2_kernel<<<grid, block, 0, stream>>>(spikes, w_out, b_out, x, alpha, out, M, H, H);
}
